// Encoder_54915451847178
// MI455X (gfx1250) — matrix operand order repaired and checked by reading, not compiled, not run on hardware
//
#include <hip/hip_runtime.h>
#include <hip/hip_bf16.h>

#define D_MODEL 512
#define N_LAYERS 6
#define N_HEADS 8
#define INNER 2048
#define SEQ 128
#define BATCH 256
#define HEAD_DIM 64
#define TOKENS (BATCH * SEQ) /* 32768 */

typedef __attribute__((ext_vector_type(16))) _Float16 v16h;
typedef __attribute__((ext_vector_type(8)))  float    v8f;
typedef _Float16 h16_t;

// ---------------------------------------------------------------------------
// Embedding lookup + sinusoidal positional encoding. Writes f32 state and the
// f16 copy used as the next GEMM's A operand.
// ---------------------------------------------------------------------------
__global__ void embed_pe_kernel(const int* __restrict__ x,
                                const float* __restrict__ emb,
                                float* __restrict__ h32,
                                h16_t* __restrict__ a16)
{
    long idx = (long)blockIdx.x * blockDim.x + threadIdx.x;
    int  d   = (int)(idx & (D_MODEL - 1));
    long bs  = idx >> 9;                 // b*SEQ + s   (D_MODEL = 2^9)
    int  s   = (int)(bs & (SEQ - 1));
    int  tok = x[bs];
    int  i   = d >> 1;
    // 10000^(-2i/512) = exp(-(2i/512)*ln(10000))
    float freq = __expf(-(float)(2 * i) * (9.210340371976184f / 512.0f));
    float th   = (float)s * freq;
    float pe   = (d & 1) ? __cosf(th) : __sinf(th);
    float v    = emb[(long)tok * D_MODEL + d] + pe;
    h32[idx] = v;
    a16[idx] = (h16_t)v;
}

// ---------------------------------------------------------------------------
// Weight repacks into f16, transposed to [N][K] (contiguous K) for the GEMM.
// ---------------------------------------------------------------------------
__global__ void pack_qkv_kernel(const float* __restrict__ W, h16_t* __restrict__ out)
{
    int idx = blockIdx.x * 256 + threadIdx.x;  // over 512*512, idx = n*512 + k
    int n = idx >> 9, k = idx & 511;
    // W is [H, D, E]; column n = h*64 + e
    out[idx] = (h16_t)W[(long)((n >> 6) * D_MODEL + k) * HEAD_DIM + (n & 63)];
}

__global__ void pack_wT_kernel(const float* __restrict__ W, h16_t* __restrict__ out,
                               int K, int N)
{
    long idx = (long)blockIdx.x * 256 + threadIdx.x;  // over K*N, idx = n*K + k
    int n = (int)(idx / K), k = (int)(idx % K);
    out[idx] = (h16_t)W[(long)k * N + n];
}

// ---------------------------------------------------------------------------
// Residual add + LayerNorm over (S, D) per batch row (65536 elems). One block
// per batch element; writes normalized f32 state and f16 GEMM operand.
// ---------------------------------------------------------------------------
__global__ void __launch_bounds__(256)
ln_residual_kernel(const float* __restrict__ h, const float* __restrict__ delta,
                   float* __restrict__ out, h16_t* __restrict__ out16)
{
    const int  b    = blockIdx.x;
    const int  tid  = threadIdx.x;
    const long base = (long)b * (SEQ * D_MODEL);
    float s = 0.f, s2 = 0.f;
    for (int i = tid; i < SEQ * D_MODEL; i += 256) {
        float v = h[base + i] + delta[base + i];
        s += v; s2 += v * v;
    }
    __shared__ float ss[256], ss2[256];
    ss[tid] = s; ss2[tid] = s2;
    __syncthreads();
    for (int st = 128; st > 0; st >>= 1) {
        if (tid < st) { ss[tid] += ss[tid + st]; ss2[tid] += ss2[tid + st]; }
        __syncthreads();
    }
    const float inv_n = 1.0f / (float)(SEQ * D_MODEL);
    float mean = ss[0] * inv_n;
    float var  = ss2[0] * inv_n - mean * mean;
    float rinv = rsqrtf(var + 1e-5f);
    for (int i = tid; i < SEQ * D_MODEL; i += 256) {
        float v = (h[base + i] + delta[base + i] - mean) * rinv;
        out[base + i]   = v;
        out16[base + i] = (h16_t)v;
    }
}

// ---------------------------------------------------------------------------
// Tiled WMMA GEMM with double-buffered LDS and async global->LDS tile copies
// (GLOBAL_LOAD_ASYNC_TO_LDS_B128, ASYNCcnt).  C[z](M,N) = op( A * B ).
//   A: f16 row-major (contiguous in K), per-z offset zo*aOuter + zi*aInner
//   B: f16; bIsKN=0 -> memory is [N][K] (contiguous K: packed weights, K^T),
//                      tiles fetched with async-to-LDS copies.
//           bIsKN=1 -> memory is [K][N] row-major (V matrix); synchronous
//                      transposing fill (must transit VGPRs).
//   Epilogue: v = acc*scale + bias[n]; optional relu; store f16 or f32.
// Fragment layouts per CDNA5 ISA 7.12.2 for V_WMMA_F32_16X16X32_F16.
// ---------------------------------------------------------------------------
template<int BM, int BN, int WMF, int WNF>
__global__ void __launch_bounds__(256)
gemm_wmma(const h16_t* __restrict__ A, int lda, long aOuter, long aInner,
          const h16_t* __restrict__ B, int ldb, long bOuter, long bInner, int bIsKN,
          float* __restrict__ C32, h16_t* __restrict__ C16, int ldc,
          long cOuter, long cInner,
          const float* __restrict__ bias, float scale, int doRelu,
          int K, int innerCnt)
{
    constexpr int BK  = 32;
    constexpr int LDT = BK + 8;                 // LDS row stride (halfs), keeps 16B align
    constexpr int WAVES_M = BM / (16 * WMF);
    constexpr int WAVES_N = BN / (16 * WNF);
    static_assert(WAVES_M * WAVES_N == 8, "need 8 waves (256 threads)");

    __shared__ __align__(16) h16_t lds_a[2][BM * LDT];
    __shared__ __align__(16) h16_t lds_b[2][BN * LDT];

    const int tid = threadIdx.x;
    const int z   = blockIdx.z;
    const int zo  = z / innerCnt, zi = z % innerCnt;
    const h16_t* Ab = A + (long)zo * aOuter + (long)zi * aInner
                        + (long)blockIdx.y * BM * lda;
    const h16_t* Bb = B + (long)zo * bOuter + (long)zi * bInner;
    const long cBase = (long)zo * cOuter + (long)zi * cInner;
    const int  n0    = blockIdx.x * BN;

    const int wave = tid >> 5, lane = tid & 31;
    const int wm = wave % WAVES_M, wn = wave / WAVES_M;
    const int ln = lane & 15, lh = lane >> 4;

    // Stage one BK-slice of A and B into LDS buffer `buf`.
    auto issue_tiles = [&](int k0, int buf) {
        // ---- A tile: BM x 32 halfs, async 16B copies straight into LDS
        for (int c = tid; c < BM * 4; c += 256) {
            int r = c >> 2, cc = c & 3;
            unsigned la = (unsigned)(size_t)&lds_a[buf][r * LDT + cc * 8];
            unsigned long long ga =
                (unsigned long long)(size_t)&Ab[(long)r * lda + k0 + cc * 8];
            asm volatile("global_load_async_to_lds_b128 %0, %1, off"
                         :: "v"(la), "v"(ga) : "memory");
        }
        // ---- B tile -> lds_b[buf][n][k]
        if (!bIsKN) {
            for (int c = tid; c < BN * 4; c += 256) {
                int r = c >> 2, cc = c & 3;
                unsigned la = (unsigned)(size_t)&lds_b[buf][r * LDT + cc * 8];
                unsigned long long ga =
                    (unsigned long long)(size_t)&Bb[(long)(n0 + r) * ldb + k0 + cc * 8];
                asm volatile("global_load_async_to_lds_b128 %0, %1, off"
                             :: "v"(la), "v"(ga) : "memory");
            }
        } else {
            for (int c = tid; c < BK * (BN / 4); c += 256) {
                int k  = c / (BN / 4);
                int nb = (c % (BN / 4)) * 4;
                union { uint2 u; h16_t h[4]; } t;
                t.u = *(const uint2*)&Bb[(long)(k0 + k) * ldb + n0 + nb];
                lds_b[buf][(nb + 0) * LDT + k] = t.h[0];
                lds_b[buf][(nb + 1) * LDT + k] = t.h[1];
                lds_b[buf][(nb + 2) * LDT + k] = t.h[2];
                lds_b[buf][(nb + 3) * LDT + k] = t.h[3];
            }
        }
    };

    v8f acc[WMF][WNF] = {};

    issue_tiles(0, 0);
    int buf = 0;
    for (int k0 = 0; k0 < K; k0 += BK) {
        // my async copies into lds[buf] done ...
        asm volatile("s_wait_asynccnt 0x0" ::: "memory");
        // ... and everyone else's too (also fences the sync-path ds_stores)
        __syncthreads();
        // prefetch next slice into the other buffer while we compute
        if (k0 + BK < K) issue_tiles(k0 + BK, buf ^ 1);

        v16h afr[WMF], bfr[WNF];
#pragma unroll
        for (int i = 0; i < WMF; ++i) {
            // A 16x32: lane m=ln, half=lh; k chunks at lh*8 and 16+lh*8
            const h16_t* pa = &lds_a[buf][(wm * WMF * 16 + i * 16 + ln) * LDT + lh * 8];
            union { v16h v; uint4 u[2]; } ua;
            ua.u[0] = *(const uint4*)pa;
            ua.u[1] = *(const uint4*)(pa + 16);
            afr[i] = ua.v;
        }
#pragma unroll
        for (int j = 0; j < WNF; ++j) {
            // B 32x16: lane n=ln, half=lh; k chunks at lh*8 and 16+lh*8
            const h16_t* pb = &lds_b[buf][(wn * WNF * 16 + j * 16 + ln) * LDT + lh * 8];
            union { v16h v; uint4 u[2]; } ub;
            ub.u[0] = *(const uint4*)pb;
            ub.u[1] = *(const uint4*)(pb + 16);
            bfr[j] = ub.v;
        }
#pragma unroll
        for (int i = 0; i < WMF; ++i)
#pragma unroll
            for (int j = 0; j < WNF; ++j)
                acc[i][j] = __builtin_amdgcn_wmma_f32_16x16x32_f16(
                    false, afr[i], false, bfr[j], (short)0, acc[i][j], false, false);
        buf ^= 1;
    }

    // ---- epilogue: D f32 layout m = r + lh*8, n = ln
#pragma unroll
    for (int i = 0; i < WMF; ++i)
#pragma unroll
        for (int j = 0; j < WNF; ++j) {
            int gm0 = blockIdx.y * BM + wm * WMF * 16 + i * 16 + lh * 8;
            int gn  = n0 + wn * WNF * 16 + j * 16 + ln;
            float bv = bias ? bias[gn] : 0.0f;
#pragma unroll
            for (int r = 0; r < 8; ++r) {
                float v = acc[i][j][r] * scale + bv;
                if (doRelu) v = fmaxf(v, 0.0f);
                long off = cBase + (long)(gm0 + r) * ldc + gn;
                if (C16) C16[off] = (h16_t)v;
                else     C32[off] = v;
            }
        }
}

// ---------------------------------------------------------------------------
extern "C" void kernel_launch(void* const* d_in, const int* in_sizes, int n_in,
                              void* d_out, int out_size, void* d_ws, size_t ws_size,
                              hipStream_t stream)
{
    (void)in_sizes; (void)n_in; (void)out_size; (void)ws_size;
    const int*   x   = (const int*)  d_in[0];
    const float* emb = (const float*)d_in[1];
    const float* Wq  = (const float*)d_in[2];
    const float* bq  = (const float*)d_in[3];
    const float* Wk  = (const float*)d_in[4];
    const float* bk  = (const float*)d_in[5];
    const float* Wv  = (const float*)d_in[6];
    const float* bv  = (const float*)d_in[7];
    const float* W1  = (const float*)d_in[8];
    const float* b1  = (const float*)d_in[9];
    const float* W2  = (const float*)d_in[10];
    const float* b2  = (const float*)d_in[11];

    const size_t MB = 1024ull * 1024ull;
    char* ws = (char*)d_ws;
    float* h32   = (float*)(ws + 0);          //  64MB  running hidden state (f32)
    float* tmp32 = (float*)(ws + 64 * MB);    //  64MB  attn-out / ffn-out (f32)
    h16_t* A16   = (h16_t*)(ws + 128 * MB);   //  32MB  f16 of h (GEMM A operand)
    h16_t* Q16   = (h16_t*)(ws + 160 * MB);   //  32MB  [B,S,H,E]
    h16_t* K16   = (h16_t*)(ws + 192 * MB);   //  32MB
    h16_t* V16   = (h16_t*)(ws + 224 * MB);   //  32MB
    h16_t* S16   = (h16_t*)(ws + 256 * MB);   //  64MB  scores [B,H,S,S]
    h16_t* F16   = (h16_t*)(ws + 256 * MB);   // 128MB  ffn hidden (reuses S16 region)
    h16_t* WqT   = (h16_t*)(ws + 384 * MB);           // 512KB each, [N=512][K=512]
    h16_t* WkT   = (h16_t*)(ws + 384 * MB + 512 * 1024);
    h16_t* WvT   = (h16_t*)(ws + 385 * MB);
    h16_t* W1T   = (h16_t*)(ws + 386 * MB);   //   2MB  [N=2048][K=512]
    h16_t* W2T   = (h16_t*)(ws + 388 * MB);   //   2MB  [N=512][K=2048]

    embed_pe_kernel<<<(TOKENS * D_MODEL) / 256, 256, 0, stream>>>(x, emb, h32, A16);

    const float scl = 0.125f;  // HEAD_DIM^-0.5

    for (int l = 0; l < N_LAYERS; ++l) {
        pack_qkv_kernel<<<(512 * 512) / 256, 256, 0, stream>>>(Wq + (long)l * 512 * 512, WqT);
        pack_qkv_kernel<<<(512 * 512) / 256, 256, 0, stream>>>(Wk + (long)l * 512 * 512, WkT);
        pack_qkv_kernel<<<(512 * 512) / 256, 256, 0, stream>>>(Wv + (long)l * 512 * 512, WvT);
        pack_wT_kernel<<<(512 * 2048) / 256, 256, 0, stream>>>(W1 + (long)l * 512 * 2048, W1T, 512, 2048);
        pack_wT_kernel<<<(512 * 2048) / 256, 256, 0, stream>>>(W2 + (long)l * 2048 * 512, W2T, 2048, 512);

        // Q/K/V projections: [32768,512] x [512,512] (+bias), f16 out
        dim3 gQKV(512 / 128, TOKENS / 128, 1);
        gemm_wmma<128, 128, 2, 4><<<gQKV, 256, 0, stream>>>(
            A16, 512, 0, 0, WqT, 512, 0, 0, 0,
            nullptr, Q16, 512, 0, 0, bq + l * 512, 1.0f, 0, 512, 1);
        gemm_wmma<128, 128, 2, 4><<<gQKV, 256, 0, stream>>>(
            A16, 512, 0, 0, WkT, 512, 0, 0, 0,
            nullptr, K16, 512, 0, 0, bk + l * 512, 1.0f, 0, 512, 1);
        gemm_wmma<128, 128, 2, 4><<<gQKV, 256, 0, stream>>>(
            A16, 512, 0, 0, WvT, 512, 0, 0, 0,
            nullptr, V16, 512, 0, 0, bv + l * 512, 1.0f, 0, 512, 1);

        // scores[b,h] = (Q K^T)*scale : 2048 batched 128x128x64 GEMMs, f16 out
        dim3 gS(1, 1, BATCH * N_HEADS);
        gemm_wmma<128, 128, 2, 4><<<gS, 256, 0, stream>>>(
            Q16, 512, (long)SEQ * 512, 64,
            K16, 512, (long)SEQ * 512, 64, 0,
            nullptr, S16, 128, (long)N_HEADS * SEQ * SEQ, (long)SEQ * SEQ,
            nullptr, scl, 0, 64, N_HEADS);

        // attn[b,h] = scores * V : 2048 batched 128x64x128 GEMMs, f32 out [B,S,H*E]
        gemm_wmma<128, 64, 2, 2><<<gS, 256, 0, stream>>>(
            S16, 128, (long)N_HEADS * SEQ * SEQ, (long)SEQ * SEQ,
            V16, 512, (long)SEQ * 512, 64, 1,
            tmp32, nullptr, 512, (long)SEQ * 512, 64,
            nullptr, 1.0f, 0, 128, N_HEADS);

        // h = LN(h + attn)
        ln_residual_kernel<<<BATCH, 256, 0, stream>>>(h32, tmp32, h32, A16);

        // FFN: relu(h W1 + b1) W2 + b2
        dim3 gF1(INNER / 128, TOKENS / 128, 1);
        gemm_wmma<128, 128, 2, 4><<<gF1, 256, 0, stream>>>(
            A16, 512, 0, 0, W1T, 512, 0, 0, 0,
            nullptr, F16, INNER, 0, 0, b1 + l * INNER, 1.0f, 1, 512, 1);

        dim3 gF2(512 / 128, TOKENS / 128, 1);
        gemm_wmma<128, 128, 2, 4><<<gF2, 256, 0, stream>>>(
            F16, INNER, 0, 0, W2T, INNER, 0, 0, 0,
            tmp32, nullptr, 512, 0, 0, b2 + l * 512, 1.0f, 0, INNER, 1);

        // h = LN(h + ffn); last layer writes d_out directly
        float* lnOut = (l == N_LAYERS - 1) ? (float*)d_out : h32;
        ln_residual_kernel<<<BATCH, 256, 0, stream>>>(h32, tmp32, lnOut, A16);
    }
}
